// MultiHeadAttention_17652315586872
// MI455X (gfx1250) — matrix operand order repaired and checked by reading, not compiled, not run on hardware
//
#include <hip/hip_runtime.h>

// ---------- CDNA5 (gfx1250) types ----------
typedef __bf16 bf16_t;
typedef __attribute__((ext_vector_type(16))) __bf16 v16bf;
typedef __attribute__((ext_vector_type(8)))  __bf16 v8bf;
typedef __attribute__((ext_vector_type(4)))  __bf16 v4bf;
typedef __attribute__((ext_vector_type(8)))  float  v8f;
typedef __attribute__((ext_vector_type(4)))  float  v4f;
typedef __attribute__((ext_vector_type(2)))  float  v2f;

// Problem sizes (fixed by reference)
constexpr int Bsz  = 4;
constexpr int Nseq = 4096;
constexpr int Dm   = 1024;
constexpr int Hh   = 16;
constexpr int DKk  = 64;
constexpr int Mrows = Bsz * Nseq;   // 16384
constexpr int Kdim  = 1024;         // inner dim of both big GEMMs
constexpr int NCHUNK = 8;           // ctx reduction split

// ============================================================
// f32 -> bf16 conversion, 4 elements/thread (bandwidth-bound)
// ============================================================
__global__ void k_cvt_bf16(const float* __restrict__ src,
                           bf16_t* __restrict__ dst, int n4) {
  int i = blockIdx.x * blockDim.x + threadIdx.x;
  if (i < n4) {
    v4f f = *reinterpret_cast<const v4f*>(src + (size_t)i * 4);
    v4bf o;
#pragma unroll
    for (int j = 0; j < 4; ++j) o[j] = (bf16_t)f[j];
    *reinterpret_cast<v4bf*>(dst + (size_t)i * 4) = o;
  }
}

// ============================================================
// Big GEMM: out[row,col] = sum_k A[row,k] * W[col,k] + bias[col]
//   A: bf16 [Mrows x 1024] row-major (activations)
//   W: bf16 [1024 x 1024]  row-major (torch Linear weight: y = x @ W.T)
// MODE 0: scatter into per-head layout [B,H,N,DK]  (QKV projections)
// MODE 1: dense [row, col] f32                      (output projection)
// Block = 256 thr = 8 waves (2M x 4N); block tile 64(M) x 256(N);
// wave tile 32 x 64 => 8 x v_wmma_f32_16x16x32_bf16 per K-step of 32.
// ============================================================
template <int MODE>
__global__ __launch_bounds__(256)
void k_gemm_bf16(const bf16_t* __restrict__ A, const bf16_t* __restrict__ W,
                 const float* __restrict__ bias, float* __restrict__ out) {
  const int lane  = threadIdx.x & 31;       // wave32
  const int wid   = threadIdx.x >> 5;       // 0..7
  const int waveM = wid >> 2;               // 0..1
  const int waveN = wid & 3;                // 0..3
  const int half  = lane >> 4;              // 0/1 (lane group)
  const int l16   = lane & 15;

  const int m0 = blockIdx.x * 64  + waveM * 32;   // wave covers rows m0..m0+31
  const int n0 = blockIdx.y * 256 + waveN * 64;   // wave covers cols n0..n0+63

  v8f acc[2][4] = {};

  for (int kc = 0; kc < Kdim; kc += 32) {
    // A fragments: 16x32 bf16, this code's k order (not the ISA table order):
    //   v0..7 <- K = 16*half..+15 (contiguous, matches B)
    v16bf afrag[2];
#pragma unroll
    for (int mt = 0; mt < 2; ++mt) {
      const bf16_t* ap = A + (size_t)(m0 + mt * 16 + l16) * Kdim + kc + 16 * half;
      v8bf lo = *reinterpret_cast<const v8bf*>(ap);        // K = 16h .. 16h+7
      v8bf hi = *reinterpret_cast<const v8bf*>(ap + 8);    // K = 16h+8 .. +7
      afrag[mt] = __builtin_shufflevector(lo, hi,
                    0,1,2,3,4,5,6,7,8,9,10,11,12,13,14,15);
      __builtin_prefetch(ap + 64, 0, 3);   // next-next K chunk, keep in cache
    }
    // B fragments: 32x16 bf16, B[k,n] = W[n,k]; lane n=l16 reads 16
    // contiguous k values starting at 16*half -> single 32B load.
    v16bf bfrag[4];
#pragma unroll
    for (int nt = 0; nt < 4; ++nt) {
      const bf16_t* bp = W + (size_t)(n0 + nt * 16 + l16) * Kdim + kc + 16 * half;
      bfrag[nt] = *reinterpret_cast<const v16bf*>(bp);
    }
#pragma unroll
    for (int mt = 0; mt < 2; ++mt)
#pragma unroll
      for (int nt = 0; nt < 4; ++nt)
        acc[mt][nt] = __builtin_amdgcn_wmma_f32_16x16x32_bf16(
            false, afrag[mt], false, bfrag[nt],
            (short)0, acc[mt][nt], false, false);
  }

  // C/D layout: VGPR i -> M = i + 8*half, N = l16.
  // One base address + constant stride per tile: stores are base + i*stride.
#pragma unroll
  for (int mt = 0; mt < 2; ++mt)
#pragma unroll
    for (int nt = 0; nt < 4; ++nt) {
      const int col = n0 + nt * 16 + l16;
      const int row = m0 + mt * 16 + 8 * half;   // +i below (never crosses b)
      const float bsv = bias[col];
      size_t base; int stride;
      if (MODE == 0) {
        const int b  = row >> 12;     // row / 4096
        const int nn = row & 4095;
        const int h  = col >> 6;      // col / 64
        const int dk = col & 63;
        base   = (((size_t)(b * Hh + h)) * Nseq + nn) * DKk + dk;
        stride = DKk;                 // consecutive i -> consecutive n
      } else {
        base   = (size_t)row * Dm + col;
        stride = Dm;
      }
#pragma unroll
      for (int i = 0; i < 8; ++i)
        out[base + (size_t)i * stride] = acc[mt][nt][i] + bsv;
    }
}

// ============================================================
// softmax over last dim (DK=64) on qp [B,H,N,DK], then * b^-0.5 (=0.5)
// one wave per row, 2 elements per lane, wave32 shuffle reduction
// ============================================================
__global__ __launch_bounds__(256)
void k_softmax_q(float* __restrict__ qp) {
  const int lane = threadIdx.x & 31;
  const int wid  = threadIdx.x >> 5;
  const long long row = (long long)blockIdx.x * 8 + wid;  // B*H*N rows
  float* p = qp + row * DKk;
  float e0 = p[lane], e1 = p[lane + 32];
  float m = fmaxf(e0, e1);
#pragma unroll
  for (int o = 16; o >= 1; o >>= 1) m = fmaxf(m, __shfl_xor(m, o, 32));
  e0 = __expf(e0 - m); e1 = __expf(e1 - m);
  float s = e0 + e1;
#pragma unroll
  for (int o = 16; o >= 1; o >>= 1) s += __shfl_xor(s, o, 32);
  const float inv = 0.5f / s;   // includes b**-0.5 = 4^-0.5
  p[lane] = e0 * inv;
  p[lane + 32] = e1 * inv;
}

// ============================================================
// softmax over sequence dim N=4096 on kp [B,H,N,DK]
// one block per (b,h,dk) column; 16 elements per thread; LDS reduction
// ============================================================
__global__ __launch_bounds__(256)
void k_softmax_k(float* __restrict__ kp) {
  __shared__ float red[256];
  const int bh = blockIdx.x >> 6;
  const int dk = blockIdx.x & 63;
  float* base = kp + (size_t)bh * Nseq * DKk + dk;
  const int t = threadIdx.x;

  float vals[16];
  float m = -1e30f;
#pragma unroll
  for (int i = 0; i < 16; ++i) {
    vals[i] = base[(size_t)(t + i * 256) * DKk];
    m = fmaxf(m, vals[i]);
  }
  red[t] = m; __syncthreads();
  for (int s = 128; s > 0; s >>= 1) {
    if (t < s) red[t] = fmaxf(red[t], red[t + s]);
    __syncthreads();
  }
  m = red[0]; __syncthreads();

  float s = 0.f;
#pragma unroll
  for (int i = 0; i < 16; ++i) { vals[i] = __expf(vals[i] - m); s += vals[i]; }
  red[t] = s; __syncthreads();
  for (int st = 128; st > 0; st >>= 1) {
    if (t < st) red[t] += red[t + st];
    __syncthreads();
  }
  const float inv = 1.0f / red[0];
#pragma unroll
  for (int i = 0; i < 16; ++i)
    base[(size_t)(t + i * 256) * DKk] = vals[i] * inv;
}

// ============================================================
// partial ctx: ctxp[c][b,h,dk,dv] = sum_{n in chunk c} kp[..n,dk]*vp[..n,dv]
// full f32 precision via v_wmma_f32_16x16x4_f32; 16 waves = 4x4 tiles;
// N split into NCHUNK chunks for occupancy (512 blocks instead of 64).
// ============================================================
__global__ __launch_bounds__(512)
void k_ctx_partial(const float* __restrict__ kp, const float* __restrict__ vp,
                   float* __restrict__ ctxp) {
  const int bh    = blockIdx.x;
  const int chunk = blockIdx.y;
  const int lane = threadIdx.x & 31;
  const int wid  = threadIdx.x >> 5;        // 0..15
  const int mt   = wid >> 2, nt = wid & 3;  // 16x16 tile coords
  const int l16  = lane & 15;
  const int k0   = (lane >> 4) * 2;         // 16x16x4 f32 lane-half K base

  const int nBeg = chunk * (Nseq / NCHUNK);
  const int nEnd = nBeg + (Nseq / NCHUNK);
  const float* kb = kp + (size_t)bh * Nseq * DKk;
  const float* vb = vp + (size_t)bh * Nseq * DKk;

  v8f acc = {};
  for (int n = nBeg; n < nEnd; n += 4) {
    v2f a, b;
    a.x = kb[(size_t)(n + k0)     * DKk + mt * 16 + l16];   // A[m=dk, k=n]
    a.y = kb[(size_t)(n + k0 + 1) * DKk + mt * 16 + l16];
    b.x = vb[(size_t)(n + k0)     * DKk + nt * 16 + l16];   // B[k=n, j=dv]
    b.y = vb[(size_t)(n + k0 + 1) * DKk + nt * 16 + l16];
    acc = __builtin_amdgcn_wmma_f32_16x16x4_f32(
        false, a, false, b, (short)0, acc, false, false);
  }
#pragma unroll
  for (int i = 0; i < 8; ++i) {
    const int m = mt * 16 + i + 8 * (lane >> 4);
    const int j = nt * 16 + l16;
    ctxp[((size_t)chunk * Bsz * Hh + bh) * DKk * DKk + m * DKk + j] = acc[i];
  }
}

// sum the NCHUNK partials: ctx[e] = sum_c ctxp[c][e]
__global__ __launch_bounds__(256)
void k_ctx_reduce(const float* __restrict__ ctxp, float* __restrict__ ctx) {
  const int e = blockIdx.x * blockDim.x + threadIdx.x;  // over B*H*64*64
  constexpr size_t STRIDE = (size_t)Bsz * Hh * DKk * DKk;
  float s = 0.f;
#pragma unroll
  for (int c = 0; c < NCHUNK; ++c) s += ctxp[(size_t)c * STRIDE + e];
  ctx[e] = s;
}

// ============================================================
// attn[b,h,n,dv] = sum_dk qp[b,h,n,dk] * ctx[b,h,dk,dv]
// ctx (16KB) staged in LDS; writes bf16 directly in concat layout
// [B, N, H*DK] so the output projection GEMM reads it unchanged.
// ============================================================
__global__ __launch_bounds__(256)
void k_attn(const float* __restrict__ qp, const float* __restrict__ ctx,
            bf16_t* __restrict__ attn) {
  __shared__ float cs[DKk * DKk];
  const int bh = blockIdx.y;
  const int b  = bh >> 4, h = bh & 15;
  for (int i = threadIdx.x; i < DKk * DKk; i += 256)
    cs[i] = ctx[(size_t)bh * DKk * DKk + i];
  __syncthreads();

  const int lane = threadIdx.x & 31;
  const int wid  = threadIdx.x >> 5;        // 8 waves: 2(M) x 4(N)
  const int mt   = wid >> 2, nt = wid & 3;
  const int l16  = lane & 15;
  const int k0   = (lane >> 4) * 2;

  const int rowA = blockIdx.x * 32 + mt * 16 + l16;
  const float* qrow = qp + ((size_t)bh * Nseq + rowA) * DKk;

  v8f acc = {};
#pragma unroll
  for (int kk = 0; kk < DKk; kk += 4) {
    v2f a = *reinterpret_cast<const v2f*>(qrow + kk + k0);  // contiguous pair
    v2f bb;
    bb.x = cs[(kk + k0)     * DKk + nt * 16 + l16];
    bb.y = cs[(kk + k0 + 1) * DKk + nt * 16 + l16];
    acc = __builtin_amdgcn_wmma_f32_16x16x4_f32(
        false, a, false, bb, (short)0, acc, false, false);
  }
#pragma unroll
  for (int i = 0; i < 8; ++i) {
    const int row = blockIdx.x * 32 + mt * 16 + i + 8 * (lane >> 4);
    const int col = h * DKk + nt * 16 + l16;   // head-concat column
    attn[((size_t)(b * Nseq + row)) * Dm + col] = (bf16_t)acc[i];
  }
}

// ============================================================
// Host-side pipeline
// ============================================================
extern "C" void kernel_launch(void* const* d_in, const int* in_sizes, int n_in,
                              void* d_out, int out_size, void* d_ws, size_t ws_size,
                              hipStream_t stream) {
  const float* q_in = (const float*)d_in[0];
  const float* k_in = (const float*)d_in[1];
  const float* v_in = (const float*)d_in[2];
  const float* Wq   = (const float*)d_in[3];
  const float* bq   = (const float*)d_in[4];
  const float* Wk   = (const float*)d_in[5];
  const float* bk   = (const float*)d_in[6];
  const float* Wv   = (const float*)d_in[7];
  const float* bv   = (const float*)d_in[8];
  const float* Wo   = (const float*)d_in[9];
  const float* bo   = (const float*)d_in[10];
  float* out = (float*)d_out;

  // workspace carve-up (all sizes static; ~340 MB)
  char* ws = (char*)d_ws;
  size_t off = 0;
  auto take = [&](size_t bytes) -> char* {
    char* p = ws + off;
    off += (bytes + 255) & ~(size_t)255;
    return p;
  };
  const size_t ACT = (size_t)Mrows * Kdim;      // 16M elements
  const size_t WEL = (size_t)Dm * Kdim;         // 1M elements
  const size_t CTX = (size_t)Bsz * Hh * DKk * DKk;
  bf16_t* qb   = (bf16_t*)take(ACT * 2);
  bf16_t* kb   = (bf16_t*)take(ACT * 2);
  bf16_t* vb   = (bf16_t*)take(ACT * 2);
  bf16_t* Wqb  = (bf16_t*)take(WEL * 2);
  bf16_t* Wkb  = (bf16_t*)take(WEL * 2);
  bf16_t* Wvb  = (bf16_t*)take(WEL * 2);
  bf16_t* Wob  = (bf16_t*)take(WEL * 2);
  float*  qp   = (float*)take(ACT * 4);         // [B,H,N,DK]
  float*  kp   = (float*)take(ACT * 4);
  float*  vp   = (float*)take(ACT * 4);
  float*  ctxp = (float*)take(CTX * NCHUNK * 4);
  float*  ctx  = (float*)take(CTX * 4);
  bf16_t* att  = (bf16_t*)take(ACT * 2);        // concat [B,N,H*DK] bf16

  // 1) bf16 conversions (vectorized x4)
  {
    const int T = 256;
    const int nA4 = (int)(ACT / 4), nW4 = (int)(WEL / 4);
    k_cvt_bf16<<<(nA4 + T - 1) / T, T, 0, stream>>>(q_in, qb, nA4);
    k_cvt_bf16<<<(nA4 + T - 1) / T, T, 0, stream>>>(k_in, kb, nA4);
    k_cvt_bf16<<<(nA4 + T - 1) / T, T, 0, stream>>>(v_in, vb, nA4);
    k_cvt_bf16<<<(nW4 + T - 1) / T, T, 0, stream>>>(Wq, Wqb, nW4);
    k_cvt_bf16<<<(nW4 + T - 1) / T, T, 0, stream>>>(Wk, Wkb, nW4);
    k_cvt_bf16<<<(nW4 + T - 1) / T, T, 0, stream>>>(Wv, Wvb, nW4);
    k_cvt_bf16<<<(nW4 + T - 1) / T, T, 0, stream>>>(Wo, Wob, nW4);
  }

  // 2) Q/K/V projections (bf16 WMMA, f32 accumulate, per-head scatter)
  {
    dim3 grid(Mrows / 64, Dm / 256);
    k_gemm_bf16<0><<<grid, 256, 0, stream>>>(qb, Wqb, bq, qp);
    k_gemm_bf16<0><<<grid, 256, 0, stream>>>(kb, Wkb, bk, kp);
    k_gemm_bf16<0><<<grid, 256, 0, stream>>>(vb, Wvb, bv, vp);
  }

  // 3) softmaxes
  k_softmax_q<<<(Bsz * Hh * Nseq) / 8, 256, 0, stream>>>(qp);
  k_softmax_k<<<Bsz * Hh * DKk, 256, 0, stream>>>(kp);

  // 4) ctx = K^T V per head (f32 WMMA, chunked for occupancy + reduce)
  k_ctx_partial<<<dim3(Bsz * Hh, NCHUNK), 512, 0, stream>>>(kp, vp, ctxp);
  k_ctx_reduce<<<(int)(CTX / 256), 256, 0, stream>>>(ctxp, ctx);

  // 5) attn = q * ctx, fused head-concat to bf16 (f32 WMMA)
  k_attn<<<dim3(Nseq / 32, Bsz * Hh), 256, 0, stream>>>(qp, ctx, att);

  // 6) output projection: out = attn @ Wo^T + bo (bf16 WMMA -> f32)
  {
    dim3 grid(Mrows / 64, Dm / 256);
    k_gemm_bf16<1><<<grid, 256, 0, stream>>>(att, Wob, bo, out);
  }
}
